// LoRAAttnProcessor_80522046865427
// MI455X (gfx1250) — matrix operand order repaired and checked by reading, not compiled, not run on hardware
//
#include <hip/hip_runtime.h>
#include <hip/hip_bf16.h>
#include <stdint.h>

typedef __attribute__((ext_vector_type(16))) __bf16 v16bf;
typedef __attribute__((ext_vector_type(8)))  float  v8f;
typedef __attribute__((ext_vector_type(4)))  unsigned int su4;
typedef __attribute__((ext_vector_type(8)))  unsigned int su8;

#define Bsz 2
#define Sseq 4096
#define Dm 640
#define Hh 8
#define HD 80
#define HDP 96
#define NTOK (Bsz * Sseq)      // 8192

// ---------------------------------------------------------------------------
// WMMA wrapper: D = A(16x32 bf16) * B(32x16 bf16) + C(16x16 f32)
// ---------------------------------------------------------------------------
static __device__ inline v8f wmma_bf16(v16bf a, v16bf b, v8f c) {
  return __builtin_amdgcn_wmma_f32_16x16x32_bf16(
      /*neg_a=*/false, a, /*neg_b=*/false, b,
      /*c_mod=*/(short)0, c, /*reuse_a=*/false, /*reuse_b=*/false);
}

union FragU { v16bf v; uint4 q[2]; };

// A fragment (16x32) from row-major [M][K] source -> two 128-bit loads.
static __device__ inline v16bf load_a_frag(const __bf16* base, int stride) {
  int lane = threadIdx.x & 31;
  const __bf16* p = base + (lane & 15) * stride + (lane >> 4) * 8;
  FragU f;
  f.q[0] = *(const uint4*)(p);        // k = hi16*8 + 0..7
  f.q[1] = *(const uint4*)(p + 16);   // k = 16 + hi16*8 + 0..7
  return f.v;
}

// B fragment (32x16) from N-major [N][K] source -> two 128-bit loads.
static __device__ inline v16bf load_bt_frag(const __bf16* base, int strideK) {
  int lane = threadIdx.x & 31;
  const __bf16* p = base + (lane & 15) * strideK + (lane >> 4) * 8;
  FragU f;
  f.q[0] = *(const uint4*)(p);
  f.q[1] = *(const uint4*)(p + 16);
  return f.v;
}

// ---------------------------------------------------------------------------
// Tensor Data Mover: 2D tile (bf16 elements) global -> LDS.
// D# group0/group1 per CDNA5 ISA 8.3/8.4; 2-group form (VADDR2/3 = NULL).
// ---------------------------------------------------------------------------
static __device__ inline void tdm_load_2d(unsigned lds_off, const void* gptr,
                                          unsigned tensor_d0, unsigned tensor_d1,
                                          unsigned tile_d0, unsigned tile_d1,
                                          unsigned long long stride0) {
  unsigned long long ga = (unsigned long long)(uintptr_t)gptr;
  su4 g0;
  g0[0] = 1u;                                   // count=1, user descriptor
  g0[1] = lds_off;                              // lds_addr (bytes)
  g0[2] = (unsigned)(ga & 0xFFFFFFFFull);       // global_addr[31:0]
  g0[3] = (unsigned)((ga >> 32) & 0x1FFFFFFull) // global_addr[56:32]
          | (2u << 30);                         // type=2 ("image")
  su8 g1;
  g1[0] = 1u << 16;                             // data_size=1 -> 2 bytes
  g1[1] = (tensor_d0 & 0xFFFFu) << 16;          // tensor_dim0[15:0]
  g1[2] = (tensor_d0 >> 16) | ((tensor_d1 & 0xFFFFu) << 16);
  g1[3] = (tensor_d1 >> 16) | (tile_d0 << 16);  // tile_dim0
  g1[4] = tile_d1;                              // tile_dim1 (tile_dim2=0)
  g1[5] = (unsigned)(stride0 & 0xFFFFFFFFull);  // tensor_dim0_stride[31:0]
  g1[6] = (unsigned)((stride0 >> 32) & 0xFFFFull);
  g1[7] = 0u;
  asm volatile("tensor_load_to_lds %0, %1" :: "s"(g0), "s"(g1) : "memory");
}

// ---------------------------------------------------------------------------
// Prep kernels
// ---------------------------------------------------------------------------
__global__ void cvt_f32_bf16(const float* __restrict__ in, __bf16* __restrict__ out, int n) {
  int i = (blockIdx.x * 256 + threadIdx.x) * 4;
  if (i < n) {
    float4 v = *(const float4*)&in[i];
    out[i + 0] = (__bf16)v.x;
    out[i + 1] = (__bf16)v.y;
    out[i + 2] = (__bf16)v.z;
    out[i + 3] = (__bf16)v.w;
  }
}

// out = W^T in bf16:  out[j*Dm + i] = W[i*Dm + j]
__global__ void cvt_f32_bf16_T(const float* __restrict__ in, __bf16* __restrict__ out) {
  int idx = blockIdx.x * 256 + threadIdx.x;
  if (idx >= Dm * Dm) return;
  int i = idx / Dm, j = idx % Dm;
  out[j * Dm + i] = (__bf16)in[idx];
}

// out = (W + down @ up)^T in bf16  (rank 4, scale = 1.0)
__global__ void fold_lora_bf16_T(const float* __restrict__ W,
                                 const float* __restrict__ down,
                                 const float* __restrict__ up,
                                 __bf16* __restrict__ out) {
  int idx = blockIdx.x * 256 + threadIdx.x;
  if (idx >= Dm * Dm) return;
  int i = idx / Dm, j = idx % Dm;
  float acc = W[idx];
#pragma unroll
  for (int r = 0; r < 4; ++r) acc += down[i * 4 + r] * up[r * Dm + j];
  out[j * Dm + i] = (__bf16)acc;
}

// ---------------------------------------------------------------------------
// GEMM: C[8192 x 640] = A[8192 x 640] (bf16) * B^T[640 x 640] (bf16, N-major)
// MODE 0: fp32 out + bias. MODE 1: bf16*premul -> [b,h,s,HDP]. MODE 2: V^T.
// Block tile 64x128x(BK=64); TDM double-buffered A/B tiles; 8 WMMAs/k-step.
// ---------------------------------------------------------------------------
template <int MODE>
__global__ __launch_bounds__(256) void gemm_bf16_kernel(
    const __bf16* __restrict__ A, const __bf16* __restrict__ Bt,
    float* __restrict__ outF, const float* __restrict__ bias,
    __bf16* __restrict__ outP, float premul) {
  __shared__ alignas(16) __bf16 ldsA[2][64 * 64];    // [M=64][K=64]
  __shared__ alignas(16) __bf16 ldsBt[2][128 * 64];  // [N=128][K=64]

  int tid  = threadIdx.x;
  int wave = tid >> 5, lane = tid & 31;
  int hi16 = lane >> 4, nlo = lane & 15;
  int wm = wave >> 2, wn = wave & 3;
  int mBase = blockIdx.x * 64, nBase = blockIdx.y * 128;

  auto issue = [&](int k0, int buf) {
    tdm_load_2d((unsigned)(uintptr_t)&ldsA[buf][0],
                &A[(size_t)mBase * Dm + k0], Dm, 1u << 20, 64, 64, Dm);
    tdm_load_2d((unsigned)(uintptr_t)&ldsBt[buf][0],
                &Bt[(size_t)nBase * Dm + k0], Dm, 1u << 20, 64, 128, Dm);
  };

  v8f c[2][2] = {};

  const int T = Dm / 64;  // 10
  if (wave == 0) issue(0, 0);
  for (int t = 0; t < T; ++t) {
    if (wave == 0) {
      if (t + 1 < T) {
        issue((t + 1) * 64, (t + 1) & 1);
        __builtin_amdgcn_s_wait_tensorcnt(2);  // current tile's 2 loads done
      } else {
        __builtin_amdgcn_s_wait_tensorcnt(0);
      }
    }
    __syncthreads();

    const __bf16* At = &ldsA[t & 1][0];
    const __bf16* Bm = &ldsBt[t & 1][0];
#pragma unroll
    for (int kc = 0; kc < 2; ++kc) {
      v16bf aF[2], bF[2];
#pragma unroll
      for (int mi = 0; mi < 2; ++mi)
        aF[mi] = load_a_frag(&At[(wm * 32 + mi * 16) * 64 + kc * 32], 64);
#pragma unroll
      for (int ni = 0; ni < 2; ++ni)
        bF[ni] = load_bt_frag(&Bm[(wn * 32 + ni * 16) * 64 + kc * 32], 64);
#pragma unroll
      for (int mi = 0; mi < 2; ++mi)
#pragma unroll
        for (int ni = 0; ni < 2; ++ni)
          c[mi][ni] = wmma_bf16(aF[mi], bF[ni], c[mi][ni]);
    }
    __syncthreads();
  }

#pragma unroll
  for (int mi = 0; mi < 2; ++mi)
#pragma unroll
    for (int ni = 0; ni < 2; ++ni)
#pragma unroll
      for (int i = 0; i < 8; ++i) {
        int gr = mBase + wm * 32 + mi * 16 + i + hi16 * 8;
        int gc = nBase + wn * 32 + ni * 16 + nlo;
        float v = c[mi][ni][i];
        if (MODE == 0) {
          outF[(size_t)gr * Dm + gc] = v + bias[gc];
        } else {
          int bb = gr >> 12, ss = gr & (Sseq - 1);
          int hh = gc / HD, hd = gc % HD;
          if (MODE == 1) {  // Q/K padded head-major
            outP[((size_t)(bb * Hh + hh) * Sseq + ss) * HDP + hd] =
                (__bf16)(v * premul);
          } else {          // V transposed [b,h,hd,s]
            outP[((size_t)(bb * Hh + hh) * HD + hd) * Sseq + ss] = (__bf16)v;
          }
        }
      }
}

// ---------------------------------------------------------------------------
// Flash attention: block = 4 waves (128 thr) for one (b,h), 64 q-rows.
// KV streamed in 64-row tiles via TDM, double-buffered. Q pre-scaled by
// softmax_scale*log2(e); softmax runs in base 2.
// ---------------------------------------------------------------------------
__global__ __launch_bounds__(128) void attn_kernel(
    const __bf16* __restrict__ Qp, const __bf16* __restrict__ Kp,
    const __bf16* __restrict__ Vt, __bf16* __restrict__ Obf) {
  __shared__ alignas(16) __bf16 ldsK[2][64 * HDP];   // [kv=64][hd=96]
  __shared__ alignas(16) __bf16 ldsVt[2][HD * 64];   // [hd=80][kv=64]
  __shared__ alignas(16) __bf16 ldsP[4][16 * 64];    // per-wave P tile

  int tid  = threadIdx.x;
  int wave = tid >> 5, lane = tid & 31;
  int hi16 = lane >> 4, nlo = lane & 15;
  int bh = blockIdx.y;
  int bb = bh >> 3, hh = bh & 7;
  int q0 = blockIdx.x * 64 + wave * 16;

  const __bf16* Qbase  = Qp + ((size_t)bh * Sseq + q0) * HDP;
  const __bf16* Kbase  = Kp + (size_t)bh * Sseq * HDP;
  const __bf16* Vtbase = Vt + (size_t)bh * HD * Sseq;

  auto issueKV = [&](int kv0, int buf) {
    tdm_load_2d((unsigned)(uintptr_t)&ldsK[buf][0],
                Kbase + (size_t)kv0 * HDP, HDP, 1u << 20, HDP, 64, HDP);
    tdm_load_2d((unsigned)(uintptr_t)&ldsVt[buf][0],
                Vtbase + kv0, 1u << 20, HD, 64, HD, Sseq);
  };

  v16bf qf[3];
#pragma unroll
  for (int kc = 0; kc < 3; ++kc) qf[kc] = load_a_frag(Qbase + kc * 32, HDP);

  v8f acc[5] = {};
  float mrow[8], lrow[8];
#pragma unroll
  for (int i = 0; i < 8; ++i) { mrow[i] = -1e30f; lrow[i] = 0.0f; }

  const int T = Sseq / 64;  // 64 tiles
  if (wave == 0) issueKV(0, 0);
  for (int t = 0; t < T; ++t) {
    if (wave == 0) {
      if (t + 1 < T) {
        issueKV((t + 1) * 64, (t + 1) & 1);
        __builtin_amdgcn_s_wait_tensorcnt(2);
      } else {
        __builtin_amdgcn_s_wait_tensorcnt(0);
      }
    }
    __syncthreads();
    const __bf16* Kt  = &ldsK[t & 1][0];
    const __bf16* Vtt = &ldsVt[t & 1][0];

    // ---- scores: S(16x64) = Qs(16x96) * K^T ; 4 N-tiles x 3 K-chunks
    v8f s[4] = {};
#pragma unroll
    for (int kc = 0; kc < 3; ++kc)
#pragma unroll
      for (int j = 0; j < 4; ++j)
        s[j] = wmma_bf16(qf[kc],
                         load_bt_frag(&Kt[(j * 16) * HDP + kc * 32], HDP),
                         s[j]);

    // ---- online softmax in base 2 (rows live across a 16-lane half)
    float p[4][8], alpha[8];
#pragma unroll
    for (int i = 0; i < 8; ++i) {
      float mx = fmaxf(fmaxf(s[0][i], s[1][i]), fmaxf(s[2][i], s[3][i]));
#pragma unroll
      for (int m = 8; m >= 1; m >>= 1) mx = fmaxf(mx, __shfl_xor(mx, m, 32));
      float mnew = fmaxf(mrow[i], mx);
      alpha[i] = exp2f(mrow[i] - mnew);
      float rs = 0.0f;
#pragma unroll
      for (int j = 0; j < 4; ++j) {
        p[j][i] = exp2f(s[j][i] - mnew);
        rs += p[j][i];
      }
#pragma unroll
      for (int m = 8; m >= 1; m >>= 1) rs += __shfl_xor(rs, m, 32);
      lrow[i] = lrow[i] * alpha[i] + rs;
      mrow[i] = mnew;
    }
#pragma unroll
    for (int j = 0; j < 5; ++j)
#pragma unroll
      for (int i = 0; i < 8; ++i) acc[j][i] *= alpha[i];

    // ---- restage P (C-layout -> A-layout) through per-wave LDS
    __bf16* P = &ldsP[wave][0];
#pragma unroll
    for (int j = 0; j < 4; ++j)
#pragma unroll
      for (int i = 0; i < 8; ++i)
        P[(i + hi16 * 8) * 64 + j * 16 + nlo] = (__bf16)p[j][i];

    // ---- O += P(16x64) * V(64x80): 2 K-chunks x 5 N-tiles
#pragma unroll
    for (int kc = 0; kc < 2; ++kc) {
      v16bf pa = load_a_frag(P + kc * 32, 64);
#pragma unroll
      for (int j = 0; j < 5; ++j)
        acc[j] = wmma_bf16(pa,
                           load_bt_frag(&Vtt[(j * 16) * 64 + kc * 32], 64),
                           acc[j]);
    }
    __syncthreads();
  }

  // ---- epilogue: O /= l, write bf16 [b,s,d]
#pragma unroll
  for (int i = 0; i < 8; ++i) {
    float inv = 1.0f / lrow[i];
    int row = q0 + i + hi16 * 8;  // s index
#pragma unroll
    for (int j = 0; j < 5; ++j) {
      int col = hh * HD + j * 16 + nlo;
      Obf[((size_t)bb * Sseq + row) * Dm + col] = (__bf16)(acc[j][i] * inv);
    }
  }
}

// ---------------------------------------------------------------------------
// Launch
// ---------------------------------------------------------------------------
extern "C" void kernel_launch(void* const* d_in, const int* in_sizes, int n_in,
                              void* d_out, int out_size, void* d_ws, size_t ws_size,
                              hipStream_t stream) {
  (void)in_sizes; (void)n_in; (void)out_size; (void)ws_size;
  const float* x      = (const float*)d_in[0];
  const float* Wq     = (const float*)d_in[1];
  const float* Wk     = (const float*)d_in[2];
  const float* Wv     = (const float*)d_in[3];
  const float* Wo     = (const float*)d_in[4];
  const float* bo     = (const float*)d_in[5];
  const float* k_down = (const float*)d_in[6];
  const float* k_up   = (const float*)d_in[7];
  const float* v_down = (const float*)d_in[8];
  const float* v_up   = (const float*)d_in[9];

  char* ws = (char*)d_ws;
  const size_t XBF_B = (size_t)NTOK * Dm * 2;              // 10.5 MB
  const size_t W_B   = (size_t)Dm * Dm * 2;                // 0.82 MB
  const size_t QP_B  = (size_t)Bsz * Hh * Sseq * HDP * 2;  // 12.6 MB (padded)
  const size_t VT_B  = (size_t)Bsz * Hh * HD * Sseq * 2;   // 10.5 MB

  __bf16* Xbf = (__bf16*)(ws);
  __bf16* WqT = (__bf16*)(ws + XBF_B);
  __bf16* WkT = (__bf16*)(ws + XBF_B + W_B);
  __bf16* WvT = (__bf16*)(ws + XBF_B + 2 * W_B);
  __bf16* WoT = (__bf16*)(ws + XBF_B + 3 * W_B);
  __bf16* Qp  = (__bf16*)(ws + XBF_B + 4 * W_B);
  __bf16* Kp  = (__bf16*)(ws + XBF_B + 4 * W_B + QP_B);
  __bf16* Vtp = (__bf16*)(ws + XBF_B + 4 * W_B + 2 * QP_B);
  __bf16* Obf = (__bf16*)(ws + XBF_B + 4 * W_B + 2 * QP_B + VT_B);

  // zero hd-pad columns (80..95) of Q/K (graph-capture-safe stream memset)
  hipMemsetAsync(Qp, 0, QP_B, stream);
  hipMemsetAsync(Kp, 0, QP_B, stream);

  // fp32 -> bf16 conversions (weights transposed) + LoRA folding
  {
    int nX = NTOK * Dm;
    cvt_f32_bf16<<<(nX / 4 + 255) / 256, 256, 0, stream>>>(x, Xbf, nX);
    int nW = Dm * Dm;
    cvt_f32_bf16_T<<<(nW + 255) / 256, 256, 0, stream>>>(Wq, WqT);
    cvt_f32_bf16_T<<<(nW + 255) / 256, 256, 0, stream>>>(Wo, WoT);
    fold_lora_bf16_T<<<(nW + 255) / 256, 256, 0, stream>>>(Wk, k_down, k_up, WkT);
    fold_lora_bf16_T<<<(nW + 255) / 256, 256, 0, stream>>>(Wv, v_down, v_up, WvT);
  }

  // Q gets softmax scale * log2(e) folded in; softmax then runs in base 2.
  const float premulQ = 0.1118033988749895f * 1.4426950408889634f;

  dim3 gGrid(NTOK / 64, Dm / 128);
  gemm_bf16_kernel<1><<<gGrid, 256, 0, stream>>>(Xbf, WqT, nullptr, nullptr, Qp, premulQ);
  gemm_bf16_kernel<1><<<gGrid, 256, 0, stream>>>(Xbf, WkT, nullptr, nullptr, Kp, 1.0f);
  gemm_bf16_kernel<2><<<gGrid, 256, 0, stream>>>(Xbf, WvT, nullptr, nullptr, Vtp, 1.0f);

  attn_kernel<<<dim3(Sseq / 64, Bsz * Hh), 128, 0, stream>>>(Qp, Kp, Vtp, Obf);

  gemm_bf16_kernel<0><<<gGrid, 256, 0, stream>>>(Obf, WoT, (float*)d_out, bo, nullptr, 0.0f);
}
